// Attention_36197984371274
// MI455X (gfx1250) — matrix operand order repaired and checked by reading, not compiled, not run on hardware
//
#include <hip/hip_runtime.h>
#include <hip/hip_bf16.h>

// MI455X (gfx1250) attention: bf16 WMMA GEMMs with fp32 accumulate.
// qkv proj -> fused per-(b,h,s) attention (TDM tile loads, sim GEMM, softmax,
// out GEMM) -> out proj.

typedef __attribute__((ext_vector_type(16))) __bf16 v16bf;
typedef __attribute__((ext_vector_type(8)))  float  v8f;
typedef __attribute__((ext_vector_type(4)))  unsigned int u32x4;
typedef __attribute__((ext_vector_type(8)))  int i32x8;
typedef __attribute__((ext_vector_type(4)))  int i32x4;

#define DIM       512
#define SSPAT     4096   // 64*64 spatial
#define HEADS     8
#define DHEAD     64
#define NBATCH    16

#if defined(__AMDGCN__) && __has_builtin(__builtin_amdgcn_tensor_load_to_lds) && \
    __has_builtin(__builtin_amdgcn_s_wait_tensorcnt)
#define USE_TDM 1
#else
#define USE_TDM 0
#endif

__device__ __forceinline__ __bf16 to_bf16(float f) {
  union { float f; unsigned u; } a; a.f = f;
  unsigned u = a.u;
  unsigned r = u + 0x7FFFu + ((u >> 16) & 1u);   // round-to-nearest-even
  union { unsigned short s; __bf16 b; } c; c.s = (unsigned short)(r >> 16);
  return c.b;
}

__device__ __forceinline__ v8f wmma_bf16(v16bf a, v16bf b, v8f c) {
  // (neg_a, A, neg_b, B, c_mod, C, reuse_a, reuse_b)
  return __builtin_amdgcn_wmma_f32_16x16x32_bf16(false, a, false, b, (short)0, c,
                                                 false, false);
}

__device__ __forceinline__ void load4(const float* p, __bf16* d) {
  const float4 v = *(const float4*)p;
  d[0] = to_bf16(v.x); d[1] = to_bf16(v.y); d[2] = to_bf16(v.z); d[3] = to_bf16(v.w);
}
__device__ __forceinline__ void load4(const __bf16* p, __bf16* d) {
  d[0] = p[0]; d[1] = p[1]; d[2] = p[2]; d[3] = p[3];
}

#if USE_TDM
// Issue one TDM descriptor: 3D tile of 2-byte elements -> LDS (tight packing).
// Element-unit dims/strides per CDNA5 ISA 8.3-8.5. Groups 2/3 carry dim2 / zeros.
// This toolchain's builtin is the 6-operand form:
//   (u32x4 g0, i32x8 g1, i32x4 g2, i32x4 g3, i32x8 extra, i32 cpol)
__device__ __forceinline__ void tdm_load_2b(
    unsigned lds_addr, unsigned long long gaddr_bytes,
    unsigned tensor_d0, unsigned tensor_d1, unsigned tensor_d2,
    unsigned tile_d0, unsigned tile_d1, unsigned tile_d2,
    unsigned long long stride0, unsigned long long stride1) {
  u32x4 g0;
  g0[0] = 1u;                                           // count=1 (valid user D#)
  g0[1] = lds_addr;                                     // lds_addr @ [63:32]
  g0[2] = (unsigned)gaddr_bytes;                        // global_addr[31:0]
  g0[3] = (unsigned)(gaddr_bytes >> 32) | (2u << 30);   // global_addr[56:32] | type=2
  i32x8 g1;
  g1[0] = (int)(1u << 16);                              // data_size=1 (2 bytes)
  g1[1] = (int)((tensor_d0 & 0xFFFFu) << 16);           // tensor_dim0[15:0] @ [63:48]
  g1[2] = (int)((tensor_d0 >> 16) | ((tensor_d1 & 0xFFFFu) << 16));
  g1[3] = (int)((tensor_d1 >> 16) | (tile_d0 << 16));   // tile_dim0 @ [127:112]
  g1[4] = (int)(tile_d1 | (tile_d2 << 16));             // tile_dim1, tile_dim2
  g1[5] = (int)(unsigned)stride0;                       // tensor_dim0_stride[31:0]
  g1[6] = (int)(((unsigned)(stride0 >> 32) & 0xFFFFu) |
                (((unsigned)stride1 & 0xFFFFu) << 16)); // s0[47:32] | s1[15:0]
  g1[7] = (int)(unsigned)(stride1 >> 16);               // tensor_dim1_stride[47:16]
  i32x4 g2;
  g2[0] = (int)tensor_d2;  // tensor_dim2
  g2[1] = 0; g2[2] = 0; g2[3] = 0;
  i32x4 g3;
  g3[0] = 0; g3[1] = 0; g3[2] = 0; g3[3] = 0;
  i32x8 gx;
  gx[0] = 0; gx[1] = 0; gx[2] = 0; gx[3] = 0;
  gx[4] = 0; gx[5] = 0; gx[6] = 0; gx[7] = 0;
  __builtin_amdgcn_tensor_load_to_lds(g0, g1, g2, g3, gx, 0);
}
#endif

// C[o, s] = sum_c W[o,c] * Xin[b,c,s]   (+ q-scale or + bias)
// QKV=true : write bf16 to outB [B][O][SSPAT], scale rows o<512 by 1/sqrt(64)
// QKV=false: write fp32 to outF [B][512][SSPAT], add bias[o]
template <typename InT, bool QKV>
__global__ __launch_bounds__(256) void proj_gemm(
    const float* __restrict__ W,     // [O][512] row-major fp32
    const InT*  __restrict__ Xin,    // [B][512][SSPAT]
    float*      __restrict__ outF,
    __bf16*     __restrict__ outB,
    const float* __restrict__ bias,
    int O) {
  __shared__ __bf16 As[64][36];      // M x Kchunk (padded)
  __shared__ __bf16 Bs[32][72];      // Kchunk x N (padded)

  const int tid  = threadIdx.x;
  const int lane = tid & 31;
  const int wv   = tid >> 5;
  const int half = lane >> 4;
  const int lq   = lane & 15;
  const int m    = wv >> 1;          // M subtile 0..3
  const int np   = (wv & 1) * 2;     // N subtile pair base (0 or 2)

  const int b     = blockIdx.z;
  const int Mbase = blockIdx.y * 64;
  const int Nbase = blockIdx.x * 64;
  const size_t xbase = (size_t)b * DIM * SSPAT;

  v8f acc0 = {};
  v8f acc1 = {};

  for (int k0 = 0; k0 < DIM; k0 += 32) {
#pragma unroll
    for (int i = 0; i < 2; ++i) {    // A tile: 64x32
      const int e  = tid + i * 256;
      const int r  = e >> 3;
      const int kq = (e & 7) * 4;
      load4(W + (size_t)(Mbase + r) * DIM + k0 + kq, &As[r][kq]);
    }
#pragma unroll
    for (int i = 0; i < 2; ++i) {    // B tile: 32x64
      const int e  = tid + i * 256;
      const int kk = e >> 4;
      const int nq = (e & 15) * 4;
      load4(Xin + xbase + (size_t)(k0 + kk) * SSPAT + Nbase + nq, &Bs[kk][nq]);
    }
    // prefetch next K chunk while this one is consumed (global_prefetch_b8)
    if (k0 + 32 < DIM) {
      __builtin_prefetch(W + (size_t)(Mbase + (tid >> 3)) * DIM + (k0 + 32) + (tid & 7) * 4, 0, 0);
      __builtin_prefetch(Xin + xbase + (size_t)(k0 + 32 + (tid >> 4)) * SSPAT + Nbase + (tid & 15) * 4, 0, 0);
    }
    __syncthreads();

    v16bf a, b0, b1;
    const int row = 16 * m + lq;
#pragma unroll
    for (int j = 0; j < 16; ++j)
      a[j] = As[row][(j & 7) + 16 * (j >> 3) + 8 * half];
    const int c0 = 16 * np + lq;
#pragma unroll
    for (int j = 0; j < 16; ++j) {
      const int kb = (j & 7) + 16 * (j >> 3) + 8 * half;
      b0[j] = Bs[kb][c0];
      b1[j] = Bs[kb][c0 + 16];
    }
    acc0 = wmma_bf16(a, b0, acc0);
    acc1 = wmma_bf16(a, b1, acc1);
    __syncthreads();
  }

#pragma unroll
  for (int t = 0; t < 2; ++t) {
    const v8f acc = t ? acc1 : acc0;
    const int col = Nbase + 16 * (np + t) + lq;
#pragma unroll
    for (int r = 0; r < 8; ++r) {
      const int ro = Mbase + 16 * m + r + 8 * half;
      float v = acc[r];
      if constexpr (QKV) {
        if (ro < DIM) v *= 0.125f;   // q * dhead^-0.5
        outB[((size_t)b * O + ro) * SSPAT + col] = to_bf16(v);
      } else {
        v += bias[ro];
        outF[((size_t)b * DIM + ro) * SSPAT + col] = v;
      }
    }
  }
}

// One workgroup per (b, h, s):
//   sim[d,e] = sum_t q[d,t]*k[e,t];  attn = softmax_e(sim);
//   out[y,d] = sum_e v[y,e]*attn[d,e];  write to ao[b, h*64+d, s*64+y]
__global__ __launch_bounds__(256) void attn_kernel(
    const __bf16* __restrict__ qkv,  // [B][1536][SSPAT]
    __bf16* __restrict__ ao) {       // [B][512][SSPAT]
  __shared__ __bf16 Qs[64][64];      // q[d][t]; reused as attn[d][e] (tight: TDM dest)
  __shared__ __bf16 Ks[64][64];      // k[e][t]
  __shared__ __bf16 Vs[64][64];      // v[e][y]  (memory layout)
  __shared__ float  Sm[64][65];      // sim fp32

  const int tid  = threadIdx.x;
  const int lane = tid & 31;
  const int wv   = tid >> 5;
  const int half = lane >> 4;
  const int lq   = lane & 15;
  const int m    = wv >> 1;
  const int np   = (wv & 1) * 2;

  const int s = blockIdx.x;
  const int h = blockIdx.y;
  const int b = blockIdx.z;

  const size_t base_q = ((size_t)b * (3 * DIM) + h * DHEAD) * SSPAT;
  const size_t base_k = base_q + (size_t)DIM * SSPAT;
  const size_t base_v = base_q + (size_t)(2 * DIM) * SSPAT;

#if USE_TDM
  if (wv == 0) {
    // Q tile: element (d,t) at qkv[base_q + d*4096 + t*64 + s]  (2B gather)
    tdm_load_2b((unsigned)(uintptr_t)&Qs[0][0],
                (unsigned long long)(uintptr_t)(qkv + base_q + s),
                /*tensor*/ 1, 64, 64, /*tile*/ 1, 64, 64,
                /*stride0*/ 64ull, /*stride1*/ 4096ull);
    // K tile: same shape at base_k
    tdm_load_2b((unsigned)(uintptr_t)&Ks[0][0],
                (unsigned long long)(uintptr_t)(qkv + base_k + s),
                1, 64, 64, 1, 64, 64, 64ull, 4096ull);
    // V tile: element (e,y) at qkv[base_v + e*4096 + s*64 + y] (128B-contig lines)
    tdm_load_2b((unsigned)(uintptr_t)&Vs[0][0],
                (unsigned long long)(uintptr_t)(qkv + base_v + s * 64),
                /*tensor*/ 64, 64, 1, /*tile*/ 64, 64, 0,
                /*stride0*/ 4096ull, /*stride1*/ 0ull);
    __builtin_amdgcn_s_wait_tensorcnt(0);
  }
#else
#pragma unroll
  for (int i = 0; i < 16; ++i) {
    const int e  = tid + i * 256;
    const int tt = e & 63;
    const int d  = e >> 6;
    Qs[d][tt] = qkv[base_q + (size_t)d * SSPAT + tt * 64 + s];
    Ks[d][tt] = qkv[base_k + (size_t)d * SSPAT + tt * 64 + s];
    Vs[d][tt] = qkv[base_v + (size_t)d * SSPAT + s * 64 + tt];  // coalesced over tt
  }
#endif
  __syncthreads();

  // ---- sim = Q * K^T (contract over t, K=64 -> 2 WMMA k-chunks) ----
  v8f acc0 = {};
  v8f acc1 = {};
#pragma unroll
  for (int kc = 0; kc < 2; ++kc) {
    const int k0 = kc * 32;
    v16bf a, b0, b1;
    const int row = 16 * m + lq;
#pragma unroll
    for (int j = 0; j < 16; ++j)
      a[j] = Qs[row][k0 + (j & 7) + 16 * (j >> 3) + 8 * half];
    const int e0 = 16 * np + lq;
#pragma unroll
    for (int j = 0; j < 16; ++j) {
      const int kb = k0 + (j & 7) + 16 * (j >> 3) + 8 * half;
      b0[j] = Ks[e0][kb];          // B[t][e] = Ks[e][t]
      b1[j] = Ks[e0 + 16][kb];
    }
    acc0 = wmma_bf16(a, b0, acc0);
    acc1 = wmma_bf16(a, b1, acc1);
  }
#pragma unroll
  for (int t = 0; t < 2; ++t) {
    const v8f acc = t ? acc1 : acc0;
    const int col = 16 * (np + t) + lq;
#pragma unroll
    for (int r = 0; r < 8; ++r)
      Sm[16 * m + r + 8 * half][col] = acc[r];
  }
  __syncthreads();

  // ---- softmax over e, attn (bf16) into Qs ----
  if (tid < 64) {
    float mx = -3.0e38f;
    for (int e = 0; e < 64; ++e) mx = fmaxf(mx, Sm[tid][e]);
    float sum = 0.f;
    for (int e = 0; e < 64; ++e) sum += __expf(Sm[tid][e] - mx);
    const float inv = 1.0f / sum;
    for (int e = 0; e < 64; ++e)
      Qs[tid][e] = to_bf16(__expf(Sm[tid][e] - mx) * inv);
  }
  __syncthreads();

  // ---- out[y,d] = sum_e V[y,e] * attn[d,e] ----
  v8f o0 = {};
  v8f o1 = {};
#pragma unroll
  for (int kc = 0; kc < 2; ++kc) {
    const int k0 = kc * 32;
    v16bf a, b0, b1;
    const int row = 16 * m + lq;     // y
#pragma unroll
    for (int j = 0; j < 16; ++j) {
      const int ka = k0 + (j & 7) + 16 * (j >> 3) + 8 * half;   // e index
      a[j] = Vs[ka][row];            // A[y][e] = Vs[e][y]
    }
    const int d0 = 16 * np + lq;
#pragma unroll
    for (int j = 0; j < 16; ++j) {
      const int kb = k0 + (j & 7) + 16 * (j >> 3) + 8 * half;
      b0[j] = Qs[d0][kb];            // B[e][d] = attn[d][e]
      b1[j] = Qs[d0 + 16][kb];
    }
    o0 = wmma_bf16(a, b0, o0);
    o1 = wmma_bf16(a, b1, o1);
  }
#pragma unroll
  for (int t = 0; t < 2; ++t) {
    const v8f acc = t ? o1 : o0;
    const int d = 16 * (np + t) + lq;
#pragma unroll
    for (int r = 0; r < 8; ++r) {
      const int y = 16 * m + r + 8 * half;
      ao[((size_t)b * DIM + h * DHEAD + d) * SSPAT + s * 64 + y] = to_bf16(acc[r]);
    }
  }
}

extern "C" void kernel_launch(void* const* d_in, const int* in_sizes, int n_in,
                              void* d_out, int out_size, void* d_ws, size_t ws_size,
                              hipStream_t stream) {
  const float* x     = (const float*)d_in[0];  // [16][512][64][64]
  const float* w_qkv = (const float*)d_in[1];  // [1536][512]
  const float* w_out = (const float*)d_in[2];  // [512][512]
  const float* b_out = (const float*)d_in[3];  // [512]
  float* out = (float*)d_out;                  // [16][512][64][64]

  __bf16* qkv = (__bf16*)d_ws;                                   // 16*1536*4096 bf16
  __bf16* ao  = (__bf16*)((char*)d_ws +
                          (size_t)NBATCH * 3 * DIM * SSPAT * sizeof(__bf16));

  // 1) qkv projection (q pre-scaled), bf16 out
  proj_gemm<float, true><<<dim3(SSPAT / 64, (3 * DIM) / 64, NBATCH), 256, 0, stream>>>(
      w_qkv, x, nullptr, qkv, nullptr, 3 * DIM);

  // 2) fused attention per (b, h, s) with TDM tile loads
  attn_kernel<<<dim3(64, HEADS, NBATCH), 256, 0, stream>>>(qkv, ao);

  // 3) output projection + bias, fp32 out
  proj_gemm<__bf16, false><<<dim3(SSPAT / 64, DIM / 64, NBATCH), 256, 0, stream>>>(
      w_out, ao, out, nullptr, b_out, DIM);
}
